// FullQuantumTransformerVQC_12077448036882
// MI455X (gfx1250) — matrix operand order repaired and checked by reading, not compiled, not run on hardware
//
#include <hip/hip_runtime.h>

typedef __attribute__((ext_vector_type(16))) _Float16 v16h;
typedef __attribute__((ext_vector_type(8)))  _Float16 v8h;
typedef __attribute__((ext_vector_type(8)))  float    v8f;

#define DEPTH 2
#define NT 128

// Workspace layout (float offsets unless noted)
#define WS_U3        0        // 20 * 8*8*2  = 2560 floats
#define WS_U5        2560     // 7 * 32*32*2 = 14336 floats
#define WS_M3        16896    // 20*3*64     = 3840 floats
#define WS_FRAG_BYTES 83200   // _Float16[7*32*32*16] = 229376 bytes (16B aligned)

// ---------------------------------------------------------------------------
// Rot(phi, theta, omega) = RZ(om) RY(th) RZ(phi); complex 2x2 as 8 floats
// ---------------------------------------------------------------------------
__device__ __forceinline__ void rot_gate(float phi, float th, float om, float g[8]) {
  float c, s, sp, cp, sm, cm;
  __sincosf(0.5f * th, &s, &c);
  __sincosf(0.5f * (phi + om), &sp, &cp);   // ep = cp - i*sp
  __sincosf(0.5f * (phi - om), &sm, &cm);   // em = cm - i*sm
  g[0] =  cp * c;  g[1] = -sp * c;          // g00 = ep*c
  g[2] = -cm * s;  g[3] = -sm * s;          // g01 = -conj(em)*s
  g[4] =  cm * s;  g[5] = -sm * s;          // g10 = em*s
  g[6] =  cp * c;  g[7] =  sp * c;          // g11 = conj(ep)*c
}

// Build one column of U (StronglyEntanglingLayers) fully in registers.
// Column j of U == evolution of basis state |j>; columns are independent.
// Wire w maps to bit (NQ-1-w). Stored interleaved re/im: Uws[(k*DIM+col)*2 +{0,1}]
template <int NQ>
__device__ void build_col(const float* __restrict__ theta, int col,
                          float* __restrict__ Uws) {
  const int DIM = 1 << NQ;
  float ar[DIM], ai[DIM];
#pragma unroll
  for (int k = 0; k < DIM; k++) { ar[k] = (k == col) ? 1.f : 0.f; ai[k] = 0.f; }
#pragma unroll
  for (int l = 0; l < DEPTH; l++) {
#pragma unroll
    for (int w = 0; w < NQ; w++) {
      const float* t3 = theta + (l * NQ + w) * 3;
      float g[8];
      rot_gate(t3[0], t3[1], t3[2], g);
      const int bit = 1 << (NQ - 1 - w);
#pragma unroll
      for (int k0 = 0; k0 < DIM; k0++) {
        if (k0 & bit) continue;
        const int k1 = k0 | bit;
        float a_r = ar[k0], a_i = ai[k0], b_r = ar[k1], b_i = ai[k1];
        ar[k0] = g[0]*a_r - g[1]*a_i + g[2]*b_r - g[3]*b_i;
        ai[k0] = g[0]*a_i + g[1]*a_r + g[2]*b_i + g[3]*b_r;
        ar[k1] = g[4]*a_r - g[5]*a_i + g[6]*b_r - g[7]*b_i;
        ai[k1] = g[4]*a_i + g[5]*a_r + g[6]*b_i + g[7]*b_r;
      }
    }
    const int r = (l % (NQ - 1)) + 1;   // PennyLane default ranges
#pragma unroll
    for (int i = 0; i < NQ; i++) {
      const int cb = 1 << (NQ - 1 - i);
      const int tb = 1 << (NQ - 1 - ((i + r) % NQ));
#pragma unroll
      for (int k = 0; k < DIM; k++) {
        if ((k & cb) && !(k & tb)) {
          const int k2 = k | tb;
          float tr = ar[k], ti = ai[k];
          ar[k] = ar[k2]; ai[k] = ai[k2];
          ar[k2] = tr;    ai[k2] = ti;
        }
      }
    }
  }
#pragma unroll
  for (int k = 0; k < DIM; k++) {
    Uws[(k * DIM + col) * 2]     = ar[k];
    Uws[(k * DIM + col) * 2 + 1] = ai[k];
  }
}

// ---------------------------------------------------------------------------
// Precompute:  U matrices -> M3q (f32) and 5q W fragments (f16, pre-swizzled
// into a contiguous-k wave32 WMMA B order: lane = 16h + n holds K' = 16h + e).
// 5q VQCs: 0..5 = attn(l,g) idx 3l+g, 6 = final.
// 3q VQCs: 0..4 stem, 5..14 ffn(l,t)=5+5l+t, 15..19 reduce.
// ---------------------------------------------------------------------------
__global__ __launch_bounds__(256) void vqc_precompute(
    const float* __restrict__ ts, const float* __restrict__ ta,
    const float* __restrict__ tf, const float* __restrict__ tr,
    const float* __restrict__ tfi, float* __restrict__ ws) {
  const int tid = threadIdx.x;
  const int wave = tid >> 5, lane = tid & 31;
  float* U3 = ws + WS_U3;
  float* U5 = ws + WS_U5;
  float* M3 = ws + WS_M3;
  _Float16* frag = (_Float16*)((char*)ws + WS_FRAG_BYTES);

  if (wave < 7) {
    // one 5-qubit VQC per wave, one column per lane
    const float* th = (wave < 6) ? (ta + wave * DEPTH * 5 * 3) : tfi;
    build_col<5>(th, lane, U5 + wave * 2048);
  } else {
    // wave 7: 20 VQCs x 8 columns = 160 tasks
    for (int it = 0; it < 5; ++it) {
      int task = lane + 32 * it;
      int v = task >> 3, col = task & 7;
      const float* th;
      if (v < 5)       th = ts + v * DEPTH * 9;
      else if (v < 15) th = tf + (v - 5) * DEPTH * 9;
      else             th = tr + (v - 15) * DEPTH * 9;
      build_col<3>(th, col, U3 + v * 128);
    }
  }
  __syncthreads();

  // M3[v][p][i][j] = sum_k z(k,p) * Re(conj(U[k][i]) U[k][j]); z bit = (2-p)
  for (int e = tid; e < 20 * 3 * 64; e += 256) {
    int v = e / 192, rem = e % 192, p = rem >> 6, ij = rem & 63;
    int i = ij >> 3, j = ij & 7;
    const float* U = U3 + v * 128;
    float acc = 0.f;
#pragma unroll
    for (int k = 0; k < 8; k++) {
      float d = U[(k*8+i)*2] * U[(k*8+j)*2] + U[(k*8+i)*2+1] * U[(k*8+j)*2+1];
      acc += ((k >> (2 - p)) & 1) ? -d : d;
    }
    M3[e] = acc;
  }

  // frag[v][kc][lane][e] = W[k = 32*kc + 16h + e][n] with n = lane&15, h = lane>>4
  // W[k][n] = M5(v, n, i = k>>5 = kc, j = k&31);  z bit = (4-n)
  for (int task = tid; task < 7 * 32; task += 256) {
    int v = task >> 5, kc = task & 31;
    const float* U = U5 + v * 2048;
    const int nread = (v == 6) ? 1 : 5;
    float cir[32], cii[32];
#pragma unroll
    for (int k = 0; k < 32; k++) {
      cir[k] = U[(k*32+kc)*2]; cii[k] = U[(k*32+kc)*2+1];
    }
    _Float16* fbase = frag + v * 16384 + kc * 512;
    for (int j = 0; j < 32; j++) {
      float d[32];
#pragma unroll
      for (int k = 0; k < 32; k++)
        d[k] = cir[k] * U[(k*32+j)*2] + cii[k] * U[(k*32+j)*2+1];
      int h = j >> 4, el = j & 15;
      for (int n = 0; n < 16; n++) {
        float val = 0.f;
        if (n < nread) {
#pragma unroll
          for (int k = 0; k < 32; k++) val += ((k >> (4 - n)) & 1) ? -d[k] : d[k];
        }
        fbase[(h * 16 + n) * 16 + el] = (_Float16)val;
      }
    }
  }
}

// ---------------------------------------------------------------------------
// 3-qubit VQC: pure per-lane VALU, M read from global with uniform addresses
// (compiler scalarizes to s_load). out_p = s0^T M_p s0, s0 real product state.
// ---------------------------------------------------------------------------
__device__ __forceinline__ void vqc3(const float* ang, const float* __restrict__ M,
                                     int nread, float* outs) {
  float c0, s0, c1, s1, c2, s2;
  __sincosf(0.5f * ang[0], &s0, &c0);
  __sincosf(0.5f * ang[1], &s1, &c1);
  __sincosf(0.5f * ang[2], &s2, &c2);
  float st[8] = { c0*c1*c2, c0*c1*s2, c0*s1*c2, c0*s1*s2,
                  s0*c1*c2, s0*c1*s2, s0*s1*c2, s0*s1*s2 };
  for (int p = 0; p < nread; p++) {
    const float* Mp = M + p * 64;
    float acc = 0.f;
#pragma unroll
    for (int i = 0; i < 8; i++) {
      float t = 0.f;
#pragma unroll
      for (int j = 0; j < 8; j++) t = fmaf(Mp[i * 8 + j], st[j], t);
      acc = fmaf(st[i], t, acc);
    }
    outs[p] = acc;
  }
}

// ---------------------------------------------------------------------------
// 5-qubit VQC via WMMA: out(16 samples, 16) += Q(16,1024) x W(1024,16)
// A chunk kc:  A[m, t] = s0[m][kc] * s0[m][t]  (scalar * fixed fragment F)
// A layout (contiguous k, same as B): lane(m,h): element e -> K=16h+e
// B fragments pre-swizzled in ws. C layout: lane n=L%16, VGPR r: M = r + 8*(L/16)
// Inner loop is software-pipelined: fragments for chunk kc+1 are produced
// BEFORE the WMMA for chunk kc issues, so the v_pk_mul/ds_load writes land in
// registers last read by the WMMA one iteration back (>4 co-exec slots away),
// avoiding the WMMA->VALU WAR hazard v_nops (ISA 7.12.1).
// ---------------------------------------------------------------------------
__device__ void vqc5(const float* ang, int vqc, int nread, float* outs,
                     _Float16* sWq, _Float16* s0h, float* sOut,
                     const _Float16* __restrict__ frag, int wave, int lane) {
  __syncthreads();  // previous users of sWq/s0h/sOut are done
  {
    const uint4* src = (const uint4*)(frag + vqc * 16384);
    uint4* dst = (uint4*)sWq;
    for (int i = threadIdx.x; i < 2048; i += NT) dst[i] = src[i];
  }
  // per-thread product state -> f16 row in LDS (row stride 40 f16 = 80B,
  // conflict-free 20-dword stride for the 16 fragment loaders)
  float c[5], s[5];
#pragma unroll
  for (int w = 0; w < 5; w++) __sincosf(0.5f * ang[w], &s[w], &c[w]);
  _Float16* myrow = s0h + (wave * 32 + lane) * 40;
#pragma unroll
  for (int i = 0; i < 32; i++) {
    float a = ((i >> 4) & 1 ? s[0] : c[0]) * ((i >> 3) & 1 ? s[1] : c[1]) *
              ((i >> 2) & 1 ? s[2] : c[2]) * ((i >> 1) & 1 ? s[3] : c[3]) *
              ((i      ) & 1 ? s[4] : c[4]);
    myrow[i] = (_Float16)a;
  }
  __syncthreads();

  const int m = lane & 15, h = lane >> 4;
  for (int t = 0; t < 2; t++) {               // two 16-sample tiles per wave
    const _Float16* srow = s0h + (wave * 32 + t * 16 + m) * 40;
    const _Float16* bptr = sWq + lane * 16;
    v8h flo = *(const v8h*)(srow + 16 * h);       // K = 16h + 0..7
    v8h fhi = *(const v8h*)(srow + 16 * h + 8);   // K = 16h + 8..15
    v16h F = __builtin_shufflevector(flo, fhi, 0,1,2,3,4,5,6,7,
                                               8,9,10,11,12,13,14,15);
    v8f acc = {0.f, 0.f, 0.f, 0.f, 0.f, 0.f, 0.f, 0.f};
    // prologue: fragments for chunk 0
    v16h A  = F * srow[0];
    v16h Bf = *(const v16h*)(bptr);
#pragma unroll
    for (int kc = 0; kc < 32; kc++) {
      v16h Acur = A;
      v16h Bcur = Bf;
      if (kc < 31) {                          // produce kc+1 before issuing kc
        A  = F * srow[kc + 1];
        Bf = *(const v16h*)(bptr + (kc + 1) * 512);
      }
      acc = __builtin_amdgcn_wmma_f32_16x16x32_f16(
          false, Acur, false, Bcur, (short)0, acc, false, false);
    }
    __syncthreads();                            // prior sOut reads complete
    if (m < nread) {
#pragma unroll
      for (int r = 0; r < 8; r++)
        sOut[wave * 128 + (r + 8 * h) * 8 + m] = acc[r];
    }
    __syncthreads();
    if (h == t) {                               // lanes owning this tile's samples
      for (int p = 0; p < nread; p++) outs[p] = sOut[wave * 128 + m * 8 + p];
    }
  }
}

// ---------------------------------------------------------------------------
// Main kernel: one sample per thread; 128 threads (4 waves) per block.
// ---------------------------------------------------------------------------
__global__ __launch_bounds__(NT) void vqc_main(
    const float* __restrict__ x, const float* __restrict__ ws,
    float* __restrict__ out) {
  __shared__ __align__(16) _Float16 sWq[32 * 32 * 16];  // 32 KB (current 5q VQC)
  __shared__ __align__(16) _Float16 s0h[4 * 32 * 40];   // 10 KB product states
  __shared__ __align__(16) float    sOut[4 * 16 * 8];   // 2 KB C staging
  const float* M3 = ws + WS_M3;
  const _Float16* frag = (const _Float16*)((const char*)ws + WS_FRAG_BYTES);
  const int tid = threadIdx.x, wave = tid >> 5, lane = tid & 31;
  const long long samp = (long long)blockIdx.x * NT + tid;

  float H[15];
  H[0] = 0.f; H[14] = 0.f;
#pragma unroll
  for (int i = 0; i < 13; i++) H[i + 1] = x[samp * 13 + i];

  // Stem: 5x Q3 (replace)
  {
    float o[3];
#pragma unroll
    for (int t = 0; t < 5; t++) {
      vqc3(&H[3 * t], M3 + t * 192, 3, o);
      H[3*t] = o[0]; H[3*t+1] = o[1]; H[3*t+2] = o[2];
    }
  }

  for (int l = 0; l < 2; l++) {
    // Quantum attention: 3 groups of Q5, all read pre-update H
    float att[15];
    for (int g = 0; g < 3; g++) {
      float ang[5];
#pragma unroll
      for (int t = 0; t < 5; t++) ang[t] = H[3 * t + g];
      float o5[5];
      vqc5(ang, 3 * l + g, 5, o5, sWq, s0h, sOut, frag, wave, lane);
#pragma unroll
      for (int t = 0; t < 5; t++) att[3 * t + g] = o5[t];
    }
#pragma unroll
    for (int i = 0; i < 15; i++) H[i] += att[i];

    // Quantum FFN: circular-shift neighbor inputs, all read pre-update H
    float add[15];
#pragma unroll
    for (int t = 0; t < 5; t++) {
      float ang3[3] = { H[3 * ((t + 4) % 5) + 2], H[3 * t + 1],
                        H[3 * ((t + 1) % 5)] };
      float o[3];
      vqc3(ang3, M3 + (5 + l * 5 + t) * 192, 3, o);
      add[3*t] = o[0]; add[3*t+1] = o[1]; add[3*t+2] = o[2];
    }
#pragma unroll
    for (int i = 0; i < 15; i++) H[i] += add[i];
  }

  // Reduce: 5x Q3 (1 readout)
  float red[5];
#pragma unroll
  for (int t = 0; t < 5; t++) {
    float o;
    vqc3(&H[3 * t], M3 + (15 + t) * 192, 1, &o);
    red[t] = o;
  }

  // Final Q5, 1 readout
  float fo[5];
  vqc5(red, 6, 1, fo, sWq, s0h, sOut, frag, wave, lane);

  out[samp] = fo[0] * 3.14159262f;   // pi - eps_f32
}

extern "C" void kernel_launch(void* const* d_in, const int* in_sizes, int n_in,
                              void* d_out, int out_size, void* d_ws, size_t ws_size,
                              hipStream_t stream) {
  const float* x   = (const float*)d_in[0];
  const float* ts  = (const float*)d_in[1];
  const float* ta  = (const float*)d_in[2];
  const float* tf  = (const float*)d_in[3];
  const float* tr  = (const float*)d_in[4];
  const float* tfi = (const float*)d_in[5];
  float* ws = (float*)d_ws;                 // needs ~313 KB of scratch
  float* out = (float*)d_out;
  const int B = in_sizes[0] / 13;           // 262144; divisible by NT

  vqc_precompute<<<1, 256, 0, stream>>>(ts, ta, tf, tr, tfi, ws);
  vqc_main<<<B / NT, NT, 0, stream>>>(x, ws, out);
}
